// SpatialAttentionLayer_45612552683794
// MI455X (gfx1250) — matrix operand order repaired and checked by reading, not compiled, not run on hardware
//
#include <hip/hip_runtime.h>
#include <hip/hip_bf16.h>
#include <math.h>

typedef _Float16 v16h __attribute__((ext_vector_type(16)));
typedef float    v8f  __attribute__((ext_vector_type(8)));

#define NJ    8192      // rows of v (= cols of x)
#define NK    100       // feature dim (= rows of x)
#define NPAD  112       // 7 tiles of 16 (col 100 = ones for Z, 101..111 = 0)
#define NT    7         // N tiles
#define JT    (NJ/32)   // 256 K-tiles of 32
#define ALPHA 0.2f

__device__ __forceinline__ float lrelu(float v) { return v > 0.0f ? v : ALPHA * v; }

// ---------------- Kernel 1: s1[j] = sum_k x[k][j]*w1[k], s2 likewise ----------------
__global__ void gat_s_kernel(const float* __restrict__ x, const float* __restrict__ w,
                             float* __restrict__ s1, float* __restrict__ s2) {
    int j = blockIdx.x * blockDim.x + threadIdx.x;   // 0..8191, coalesced over j
    float a = 0.0f, b = 0.0f;
    #pragma unroll 4
    for (int k = 0; k < NK; ++k) {
        float xv = x[(size_t)k * NJ + j];
        a = fmaf(xv, w[k], a);
        b = fmaf(xv, w[NK + k], b);
    }
    s1[j] = a;
    s2[j] = b;
}

// ---------------- Kernel 2: m2 = max_j s2[j] (single block) ----------------
__global__ void gat_max_kernel(const float* __restrict__ s2, float* __restrict__ m2) {
    __shared__ float red[256];
    int tid = threadIdx.x;
    float m = -INFINITY;
    for (int j = tid; j < NJ; j += 256) m = fmaxf(m, s2[j]);
    red[tid] = m;
    __syncthreads();
    for (int off = 128; off > 0; off >>= 1) {
        if (tid < off) red[tid] = fmaxf(red[tid], red[tid + off]);
        __syncthreads();
    }
    if (tid == 0) *m2 = red[0];
}

// ---------------- Kernel 3: pre-swizzle B operand (v as f16 in WMMA B layout) -------
// B tile for K-tile jt, N-tile nt: lane L holds column n = nt*16 + (L&15),
// halves e=0..15 -> K = (L>=16 ? 8 : 0) + (e&7) + (e>=8 ? 16 : 0)  (VGPR r = e/2, packed pairs).
// Column 100 is all-ones (computes softmax denominator Z), 101..111 are zero.
__global__ void gat_pack_kernel(const float* __restrict__ x, _Float16* __restrict__ B) {
    int t    = blockIdx.x * blockDim.x + threadIdx.x;   // 256*7*32 = 57344 threads
    int lane = t & 31;
    int nt   = (t >> 5) % NT;
    int jt   = t / (32 * NT);
    int n    = nt * 16 + (lane & 15);
    int j0   = jt * 32 + ((lane >> 4) << 3);            // +8 for hi half-wave
    v16h vv;
    if (n < NK) {
        #pragma unroll
        for (int e = 0; e < 16; ++e)
            vv[e] = (_Float16)x[(size_t)n * NJ + j0 + (e & 7) + ((e & 8) << 1)];   // v[j][n] = x[n][j]
    } else {
        _Float16 fill = (n == NK) ? (_Float16)1.0f : (_Float16)0.0f;
        #pragma unroll
        for (int e = 0; e < 16; ++e) vv[e] = fill;
    }
    *(v16h*)(B + ((size_t)(jt * NT + nt) * 32 + lane) * 16) = vv;
}

// ---------------- Kernel 4: flash-style fused softmax @ v via WMMA ----------------
// blockDim = 128 (4 waves), each wave owns 32 output rows (two 16-row WMMA tiles).
__global__ void __launch_bounds__(128)
gat_attn_kernel(const float* __restrict__ s1, const float* __restrict__ s2,
                const float* __restrict__ m2p, const _Float16* __restrict__ B,
                float* __restrict__ out) {
    const int lane  = threadIdx.x & 31;
    const int wave  = threadIdx.x >> 5;
    const int i0    = (blockIdx.x * 4 + wave) * 32;  // first of 32 rows for this wave
    const int li    = lane & 15;
    const int hi    = lane >> 4;
    const int kbase = hi * 8;                        // A-layout K base for this half

    const float m2  = *m2p;
    const float s1a = s1[i0 + li];                   // A rows: both halves hold M=0..15
    const float s1b = s1[i0 + 16 + li];
    const float ma  = lrelu(s1a + m2);               // exact row max (lrelu monotone)
    const float mb  = lrelu(s1b + m2);

    v8f acc[2][NT];
    #pragma unroll
    for (int t = 0; t < 2; ++t)
        #pragma unroll
        for (int nt = 0; nt < NT; ++nt) acc[t][nt] = (v8f)0.0f;

    for (int jt = 0; jt < JT; ++jt) {
        const int j0 = jt * 32;
        float sv0[8], sv1[8];
        #pragma unroll
        for (int e = 0; e < 8; ++e) {                // two contiguous 32B runs of s2
            sv0[e] = s2[j0 + kbase + e];
            sv1[e] = s2[j0 + 16 + kbase + e];
        }
        // Synthesize P tiles in registers, exact 16-bit A layout:
        // elem e   -> K = kbase + e        (lo run)
        // elem e+8 -> K = 16 + kbase + e   (hi run)
        v16h A0, A1;
        #pragma unroll
        for (int e = 0; e < 8; ++e) {
            A0[e]     = (_Float16)__expf(lrelu(s1a + sv0[e]) - ma);
            A0[e + 8] = (_Float16)__expf(lrelu(s1a + sv1[e]) - ma);
            A1[e]     = (_Float16)__expf(lrelu(s1b + sv0[e]) - mb);
            A1[e + 8] = (_Float16)__expf(lrelu(s1b + sv1[e]) - mb);
        }
        const _Float16* bp = B + (size_t)jt * NT * 512 + (size_t)lane * 16;
        #pragma unroll
        for (int nt = 0; nt < NT; ++nt) {
            v16h Bv = *(const v16h*)(bp + (size_t)nt * 512);
            acc[0][nt] = __builtin_amdgcn_wmma_f32_16x16x32_f16(
                false, A0, false, Bv, (short)0, acc[0][nt], false, false);
            acc[1][nt] = __builtin_amdgcn_wmma_f32_16x16x32_f16(
                false, A1, false, Bv, (short)0, acc[1][nt], false, false);
        }
    }

    // Epilogue: Z_i sits at column n=100 -> n-tile 6, N=4 -> lane 4|(lane&16).
    // C layout: lane L, VGPR r holds row m = r + 8*(L>>4); same half holds its Z.
    const int zlane = 4 | (lane & 16);
    #pragma unroll
    for (int t = 0; t < 2; ++t) {
        #pragma unroll
        for (int r = 0; r < 8; ++r) {
            float z    = __shfl(acc[t][6][r], zlane, 32);
            float invz = __frcp_rn(z);               // z >= 1 (max term contributes 1)
            int row    = i0 + t * 16 + r + 8 * hi;
            #pragma unroll
            for (int nt = 0; nt < NT; ++nt) {
                int n = nt * 16 + li;
                if (n < NK) out[(size_t)row * NK + n] = acc[t][nt][r] * invz;
            }
        }
    }
}

extern "C" void kernel_launch(void* const* d_in, const int* in_sizes, int n_in,
                              void* d_out, int out_size, void* d_ws, size_t ws_size,
                              hipStream_t stream) {
    const float* x = (const float*)d_in[0];   // (100, 8192) f32
    const float* w = (const float*)d_in[1];   // (200, 1)  f32
    float* out = (float*)d_out;               // (8192, 100) f32

    // workspace layout
    float* s1 = (float*)d_ws;                                   // 8192 f32
    float* s2 = s1 + NJ;                                        // 8192 f32
    float* m2 = s2 + NJ;                                        // 1 f32
    _Float16* Bswz = (_Float16*)((char*)d_ws + 65600);          // 256*7*32*16 halves (~1.75MB)

    gat_s_kernel   <<<NJ / 256, 256, 0, stream>>>(x, w, s1, s2);
    gat_max_kernel <<<1,        256, 0, stream>>>(s2, m2);
    gat_pack_kernel<<<(JT * NT * 32) / 256, 256, 0, stream>>>(x, Bswz);
    gat_attn_kernel<<<NJ / 128, 128, 0, stream>>>(s1, s2, m2, Bswz, out);
}
